// MoE_28329604284811
// MI455X (gfx1250) — matrix operand order repaired and checked by reading, not compiled, not run on hardware
//
#include <hip/hip_runtime.h>
#include <cstddef>

// ---------------- problem constants ----------------
constexpr int DIMC   = 2048;
constexpr int TC     = 4096;   // B*S tokens
constexpr int EC     = 64;     // routed experts
constexpr int KC     = 6;      // activated experts
constexpr int GC     = 8;      // expert groups
constexpr int LGC    = 4;      // limited groups
constexpr int CAPC   = 768;    // per-expert capacity
constexpr int INTERC = 512;
constexpr int SINTERC= 1024;
constexpr float ROUTE_SCALE = 2.5f;

// ---------------- types ----------------
typedef float        v8f   __attribute__((ext_vector_type(8)));
typedef __bf16       v16bf __attribute__((ext_vector_type(16)));
typedef unsigned int v4u   __attribute__((ext_vector_type(4)));

union FragU { v16bf bf; v4u q[2]; };

__device__ __forceinline__ unsigned short f2bf(float f) {
    unsigned u = __float_as_uint(f);
    unsigned r = u + 0x7FFFu + ((u >> 16) & 1u);   // round-to-nearest-even
    return (unsigned short)(r >> 16);
}
__device__ __forceinline__ float bf2f(unsigned short h) {
    return __uint_as_float(((unsigned)h) << 16);
}
__device__ __forceinline__ v8f vzero() {
    v8f z;
#pragma unroll
    for (int i = 0; i < 8; ++i) z[i] = 0.f;
    return z;
}

// A fragment 16x32 bf16 from LDS (row-major, stride%8==0):
// lanes 0-15: M=lane, K={0..7,16..23}; lanes 16-31: M=lane-16, K={8..15,24..31}
__device__ __forceinline__ v16bf ld_a(const unsigned short* base, int stride,
                                      int mtile, int kbase, int lane) {
    int half = lane >> 4;
    int row  = mtile + (lane & 15);
    const unsigned short* p = base + row * stride + kbase + half * 8;
    FragU u;
    u.q[0] = *(const v4u*)(p);
    u.q[1] = *(const v4u*)(p + 16);
    return u.bf;
}
// B fragment 32x16 bf16 from LDS stored [N][K]:
// lanes 0-15: N=lane, K={0..7,16..23}; lanes 16-31: N=lane-16, K={8..15,24..31}
__device__ __forceinline__ v16bf ld_b(const unsigned short* base, int stride,
                                      int n0, int kbase, int lane) {
    int half = lane >> 4;
    int col  = n0 + (lane & 15);
    const unsigned short* p = base + col * stride + kbase + half * 8;
    FragU u;
    u.q[0] = *(const v4u*)(p);
    u.q[1] = *(const v4u*)(p + 16);
    return u.bf;
}

// ---------------- weight pre-pass: fp32 [R][C] -> bf16 [C][R] ----------------
__global__ __launch_bounds__(256) void transpose_cvt_kernel(const float* __restrict__ src,
                                                            unsigned short* __restrict__ dst,
                                                            int R, int C) {
    __shared__ unsigned short tile[64 * 72];
    const int b  = blockIdx.z;
    const float* s = src + (size_t)b * R * C;
    unsigned short* d = dst + (size_t)b * R * C;
    const int rb = blockIdx.y * 64, cb = blockIdx.x * 64;
    for (int i = threadIdx.x; i < 64 * 64; i += 256) {
        int r = i >> 6, c = i & 63;
        tile[c * 72 + r] = f2bf(s[(size_t)(rb + r) * C + cb + c]);
    }
    __syncthreads();
    unsigned* d32 = (unsigned*)d;
    for (int i = threadIdx.x; i < 64 * 32; i += 256) {
        int rr = i >> 5, cc = (i & 31) * 2;
        unsigned lo = tile[rr * 72 + cc], hi = tile[rr * 72 + cc + 1];
        d32[((size_t)(cb + rr) * R + rb + cc) >> 1] = lo | (hi << 16);
    }
}

// ---------------- kernel: gate scores = sigmoid(x @ gate_w^T) ----------------
__global__ __launch_bounds__(256) void gate_kernel(const float* __restrict__ x,
                                                   const float* __restrict__ gw,
                                                   float* __restrict__ scores) {
    extern __shared__ char smem[];
    float* xs = (float*)smem;            // [64][132]
    float* gs = xs + 64 * 132;           // [64][132]
    const int tid = threadIdx.x;
    const int t0  = blockIdx.x * 64;
    const int tok = tid & 63;
    const int e0  = (tid >> 6) * 16;
    float acc[16];
#pragma unroll
    for (int j = 0; j < 16; ++j) acc[j] = 0.f;

    for (int k0 = 0; k0 < DIMC; k0 += 128) {
        __syncthreads();
        for (int i = tid; i < 64 * 128; i += 256) {
            int r = i >> 7, c = i & 127;
            xs[r * 132 + c] = x[(size_t)(t0 + r) * DIMC + k0 + c];
            gs[r * 132 + c] = gw[(size_t)r * DIMC + k0 + c];
        }
        __syncthreads();
        for (int kk = 0; kk < 128; ++kk) {
            float xv = xs[tok * 132 + kk];
#pragma unroll
            for (int j = 0; j < 16; ++j)
                acc[j] += xv * gs[(e0 + j) * 132 + kk];
        }
    }
#pragma unroll
    for (int j = 0; j < 16; ++j)
        scores[(size_t)(t0 + tok) * EC + e0 + j] = 1.f / (1.f + __expf(-acc[j]));
}

// ---------------- kernel: group-limited top-k routing ----------------
__global__ void route_kernel(const float* __restrict__ scores,
                             int* __restrict__ idxb, float* __restrict__ wtsb) {
    int t = blockIdx.x * blockDim.x + threadIdx.x;
    if (t >= TC) return;
    float s[EC];
    for (int e = 0; e < EC; ++e) s[e] = scores[(size_t)t * EC + e];

    float gsv[GC];
    for (int g = 0; g < GC; ++g) {
        float m = s[g * 8];
        for (int j = 1; j < 8; ++j) m = fmaxf(m, s[g * 8 + j]);
        gsv[g] = m;
    }
    unsigned gm = 0;
    for (int it = 0; it < LGC; ++it) {
        int best = 0; float bv = -3.4e38f;
        for (int g = 0; g < GC; ++g)
            if (!((gm >> g) & 1u) && gsv[g] > bv) { bv = gsv[g]; best = g; }
        gm |= 1u << best;
    }
    float ms[EC];
    for (int e = 0; e < EC; ++e)
        ms[e] = ((gm >> (e >> 3)) & 1u) ? s[e] : 0.f;

    float sum = 0.f;
    int   sel[KC];
    for (int it = 0; it < KC; ++it) {
        int best = 0; float bv = -3.4e38f;
        for (int e = 0; e < EC; ++e)
            if (ms[e] > bv) { bv = ms[e]; best = e; }
        ms[best] = -3.4e38f;
        sel[it] = best;
        sum += s[best];
    }
    float inv = ROUTE_SCALE / sum;
    for (int it = 0; it < KC; ++it) {
        idxb[t * KC + it] = sel[it];
        wtsb[t * KC + it] = s[sel[it]] * inv;
    }
}

// ---------------- dispatch ----------------
__global__ void zero_cnt_kernel(int* cnt) { if (threadIdx.x < EC) cnt[threadIdx.x] = 0; }

__global__ void pair_kernel(const int* __restrict__ idxb, int* __restrict__ cnt,
                            int* __restrict__ rowmap, int* __restrict__ pairpos) {
    int p = blockIdx.x * blockDim.x + threadIdx.x;
    if (p >= TC * KC) return;
    int e = idxb[p];
    int pos = atomicAdd(&cnt[e], 1);
    if (pos < CAPC) {
        rowmap[e * CAPC + pos] = p / KC;
        pairpos[p] = pos;
    } else {
        pairpos[p] = -1;
    }
}

// ---------------- fused SwiGLU MLP (WMMA bf16, pre-transposed bf16 weights) ----
// Wave tiling: 16 waves; each wave owns a 32-row x 64-col tile:
//   2 A fragments x 4 B fragments; every B fragment feeds 2 WMMAs.
// LDS layout (dynamic):
//   rowidx : int[64]              @ 0       (256 B)
//   As     : ushort[64][72]       @ 256     (9216 B)
//   Ws1    : ushort[512][72]      @ 9472    (73728 B)   [N][K] bf16
//   Ws3    : ushort[512][72]      @ 83200   (73728 B)
//   Hs     : ushort[64][INTER+8]  @ 156928
template<int INTERSZ, bool GATHER>
__global__ __launch_bounds__(512) void mlp_kernel(const float* __restrict__ x,
                                                  const unsigned short* __restrict__ wAt,
                                                  const unsigned short* __restrict__ wBt,
                                                  const unsigned short* __restrict__ wCt,
                                                  const int* __restrict__ rowmap,
                                                  const int* __restrict__ cnt,
                                                  unsigned short* __restrict__ outh,
                                                  float* __restrict__ outf) {
    extern __shared__ char smem[];
    int*            rowidx = (int*)smem;
    unsigned short* As  = (unsigned short*)(smem + 256);
    unsigned short* Ws1 = (unsigned short*)(smem + 9472);
    unsigned short* Ws3 = (unsigned short*)(smem + 83200);
    unsigned short* Hs  = (unsigned short*)(smem + 156928);
    constexpr int SA = 72, SK = 72, SH = INTERSZ + 8;

    const int tid   = threadIdx.x;
    const int lane  = tid & 31;
    const int wv    = tid >> 5;           // 16 waves
    const int mrow0 = (wv & 1) * 32;      // 2 row-pairs (rows 0-31 / 32-63)
    const int ncol0 = (wv >> 1) * 64;     // 8 N strips of 64
    const int e     = GATHER ? blockIdx.y : 0;
    const int tilebase = blockIdx.x * 64;

    int rows_valid;
    const unsigned short *pA, *pB, *pC;
    if constexpr (GATHER) {
        rows_valid = cnt[e];
        if (rows_valid > CAPC) rows_valid = CAPC;
        if (tilebase >= rows_valid) return;   // uniform per block
        pA = wAt + (size_t)e * INTERSZ * DIMC;    // [INTER][DIM]
        pB = wBt + (size_t)e * INTERSZ * DIMC;
        pC = wCt + (size_t)e * DIMC * INTERSZ;    // [DIM][INTER]
    } else {
        rows_valid = TC;
        pA = wAt; pB = wBt; pC = wCt;
    }

    if (tid < 64) {
        int slot = tilebase + tid;
        int r;
        if constexpr (GATHER) r = (slot < rows_valid) ? rowmap[e * CAPC + slot] : -1;
        else                  r = slot;
        rowidx[tid] = r;
    }
    __syncthreads();

    const int half8 = (lane >> 4) * 8;
    const int nlo   = lane & 15;

    // -------- stage 1: H = silu(A@w1) * (A@w3), 512 N-columns per pass --------
    constexpr int NH = INTERSZ / 512;
    for (int nh = 0; nh < NH; ++nh) {
        // acc[mat][mi*4 + ti] : mat in {w1,w3}, mi in {0,1} rows, ti in 0..3 cols
        v8f acc[2][8];
#pragma unroll
        for (int w2i = 0; w2i < 2; ++w2i)
#pragma unroll
            for (int t = 0; t < 8; ++t) acc[w2i][t] = vzero();

        for (int k0 = 0; k0 < DIMC; k0 += 64) {
            __syncthreads();
            // A tile: 64 rows x 64 k, fp32 -> packed bf16
            for (int i = tid; i < 64 * 16; i += 512) {
                int r = i >> 4, c4 = (i & 15) * 4;
                int row = rowidx[r];
                float4 v = make_float4(0.f, 0.f, 0.f, 0.f);
                if (row >= 0) v = *(const float4*)(x + (size_t)row * DIMC + k0 + c4);
                unsigned lo = (unsigned)f2bf(v.x) | ((unsigned)f2bf(v.y) << 16);
                unsigned hi = (unsigned)f2bf(v.z) | ((unsigned)f2bf(v.w) << 16);
                *(unsigned*)(As + r * SA + c4)     = lo;
                *(unsigned*)(As + r * SA + c4 + 2) = hi;
            }
            // W1/W3 tiles: pure bf16 b128 copies, [N][K] layout
            for (int i = tid; i < 2 * 512 * 8; i += 512) {
                int mat = i >> 12;
                int ii  = i & 4095;
                int n = ii >> 3, j = ii & 7;
                const unsigned short* src =
                    (mat ? pB : pA) + (size_t)(nh * 512 + n) * DIMC + k0 + j * 8;
                unsigned short* dstW = (mat ? Ws3 : Ws1) + n * SK + j * 8;
                *(v4u*)dstW = *(const v4u*)src;
                if (k0 + 64 < DIMC) __builtin_prefetch(src + 64, 0, 1);
            }
            __syncthreads();
#pragma unroll
            for (int kk = 0; kk < 64; kk += 32) {
                v16bf af0 = ld_a(As, SA, mrow0,      kk, lane);
                v16bf af1 = ld_a(As, SA, mrow0 + 16, kk, lane);
#pragma unroll
                for (int ti = 0; ti < 4; ++ti) {
                    v16bf b1 = ld_b(Ws1, SK, ncol0 + ti * 16, kk, lane);
                    acc[0][ti]     = __builtin_amdgcn_wmma_f32_16x16x32_bf16(
                        false, af0, false, b1, (short)0, acc[0][ti],     false, false);
                    acc[0][4 + ti] = __builtin_amdgcn_wmma_f32_16x16x32_bf16(
                        false, af1, false, b1, (short)0, acc[0][4 + ti], false, false);
                    v16bf b3 = ld_b(Ws3, SK, ncol0 + ti * 16, kk, lane);
                    acc[1][ti]     = __builtin_amdgcn_wmma_f32_16x16x32_bf16(
                        false, af0, false, b3, (short)0, acc[1][ti],     false, false);
                    acc[1][4 + ti] = __builtin_amdgcn_wmma_f32_16x16x32_bf16(
                        false, af1, false, b3, (short)0, acc[1][4 + ti], false, false);
                }
            }
        }
        __syncthreads();
        // SwiGLU -> Hs (bf16)
#pragma unroll
        for (int mi = 0; mi < 2; ++mi)
#pragma unroll
            for (int ti = 0; ti < 4; ++ti)
#pragma unroll
                for (int r = 0; r < 8; ++r) {
                    int idx = mi * 4 + ti;
                    float a1 = acc[0][idx][r], a3 = acc[1][idx][r];
                    float h  = (a1 / (1.f + __expf(-a1))) * a3;
                    int m = mrow0 + mi * 16 + r + half8;
                    int n = nh * 512 + ncol0 + ti * 16 + nlo;
                    Hs[m * SH + n] = f2bf(h);
                }
    }
    __syncthreads();

    // -------- stage 2: out = H @ w2, ping-pong W buffers --------
    unsigned short* WB[2] = { Ws1, Ws3 };
    for (int nc = 0; nc < DIMC; nc += 512) {
        v8f acc[8];
#pragma unroll
        for (int t = 0; t < 8; ++t) acc[t] = vzero();

        auto stageW2 = [&](unsigned short* dstW, int kb) {
            for (int i = tid; i < 512 * 8; i += 512) {
                int n = i >> 3, j = i & 7;
                const unsigned short* src =
                    pC + (size_t)(nc + n) * INTERSZ + kb + j * 8;
                *(v4u*)(dstW + n * SK + j * 8) = *(const v4u*)src;
            }
        };
        stageW2(WB[0], 0);
        __syncthreads();
        for (int k0 = 0; k0 < INTERSZ; k0 += 64) {
            int cur = (k0 >> 6) & 1;
            if (k0 + 64 < INTERSZ) stageW2(WB[cur ^ 1], k0 + 64);
#pragma unroll
            for (int kk = 0; kk < 64; kk += 32) {
                v16bf af0 = ld_a(Hs, SH, mrow0,      k0 + kk, lane);
                v16bf af1 = ld_a(Hs, SH, mrow0 + 16, k0 + kk, lane);
#pragma unroll
                for (int ti = 0; ti < 4; ++ti) {
                    v16bf bf = ld_b(WB[cur], SK, ncol0 + ti * 16, kk, lane);
                    acc[ti]     = __builtin_amdgcn_wmma_f32_16x16x32_bf16(
                        false, af0, false, bf, (short)0, acc[ti],     false, false);
                    acc[4 + ti] = __builtin_amdgcn_wmma_f32_16x16x32_bf16(
                        false, af1, false, bf, (short)0, acc[4 + ti], false, false);
                }
            }
            __syncthreads();
        }
#pragma unroll
        for (int mi = 0; mi < 2; ++mi)
#pragma unroll
            for (int ti = 0; ti < 4; ++ti)
#pragma unroll
                for (int r = 0; r < 8; ++r) {
                    int idx = mi * 4 + ti;
                    int m   = mrow0 + mi * 16 + r + half8;
                    int col = nc + ncol0 + ti * 16 + nlo;
                    if constexpr (GATHER)
                        outh[((size_t)(e * CAPC + tilebase + m)) * DIMC + col] =
                            f2bf(acc[idx][r]);
                    else
                        outf[((size_t)(tilebase + m)) * DIMC + col] = acc[idx][r];
                }
    }
}

// ---------------- combine (gather, deterministic) ----------------
__global__ __launch_bounds__(256) void combine_kernel(const int* __restrict__ idxb,
                                                      const int* __restrict__ pairpos,
                                                      const float* __restrict__ wtsb,
                                                      const unsigned short* __restrict__ eout,
                                                      float* __restrict__ out) {
    int t = blockIdx.x;
    int ee[KC], pp[KC];
    float ww[KC];
#pragma unroll
    for (int k = 0; k < KC; ++k) {
        int p = t * KC + k;
        ee[k] = idxb[p]; pp[k] = pairpos[p]; ww[k] = wtsb[p];
    }
    for (int c = threadIdx.x; c < DIMC; c += 256) {
        float a = out[(size_t)t * DIMC + c];
#pragma unroll
        for (int k = 0; k < KC; ++k)
            if (pp[k] >= 0)
                a += ww[k] * bf2f(eout[((size_t)ee[k] * CAPC + pp[k]) * DIMC + c]);
        out[(size_t)t * DIMC + c] = a;
    }
}

// ---------------- host ----------------
static constexpr int GATE_LDS  = 2 * 64 * 132 * 4;                   // 67584
static constexpr int MLP_LDS_D = 156928 + 64 * (INTERC  + 8) * 2;    // 223488
static constexpr int MLP_LDS_S = 156928 + 64 * (SINTERC + 8) * 2;    // 289024

extern "C" void kernel_launch(void* const* d_in, const int* in_sizes, int n_in,
                              void* d_out, int out_size, void* d_ws, size_t ws_size,
                              hipStream_t stream) {
    const float* x      = (const float*)d_in[0];
    const float* gate_w = (const float*)d_in[1];
    const float* w1     = (const float*)d_in[2];
    const float* w2     = (const float*)d_in[3];
    const float* w3     = (const float*)d_in[4];
    const float* ws1    = (const float*)d_in[5];
    const float* ws2    = (const float*)d_in[6];
    const float* ws3    = (const float*)d_in[7];
    float* out = (float*)d_out;

    char* wsb = (char*)d_ws;
    size_t off = 0;
    auto alloc = [&](size_t bytes) {
        void* p = wsb + off;
        off = (off + bytes + 255) & ~(size_t)255;
        return p;
    };
    float* scores  = (float*)alloc((size_t)TC * EC * sizeof(float));
    int*   idxb    = (int*)  alloc((size_t)TC * KC * sizeof(int));
    float* wtsb    = (float*)alloc((size_t)TC * KC * sizeof(float));
    int*   cnt     = (int*)  alloc((size_t)EC * sizeof(int));
    int*   rowmap  = (int*)  alloc((size_t)EC * CAPC * sizeof(int));
    int*   pairpos = (int*)  alloc((size_t)TC * KC * sizeof(int));
    unsigned short* w1t  = (unsigned short*)alloc((size_t)EC * INTERC * DIMC * 2);
    unsigned short* w3t  = (unsigned short*)alloc((size_t)EC * INTERC * DIMC * 2);
    unsigned short* w2t  = (unsigned short*)alloc((size_t)EC * DIMC * INTERC * 2);
    unsigned short* ws1t = (unsigned short*)alloc((size_t)SINTERC * DIMC * 2);
    unsigned short* ws3t = (unsigned short*)alloc((size_t)SINTERC * DIMC * 2);
    unsigned short* ws2t = (unsigned short*)alloc((size_t)DIMC * SINTERC * 2);
    unsigned short* eout = (unsigned short*)alloc((size_t)EC * CAPC * DIMC * 2);

    (void)hipFuncSetAttribute((const void*)gate_kernel,
                              hipFuncAttributeMaxDynamicSharedMemorySize, GATE_LDS);
    (void)hipFuncSetAttribute((const void*)(mlp_kernel<INTERC, true>),
                              hipFuncAttributeMaxDynamicSharedMemorySize, MLP_LDS_D);
    (void)hipFuncSetAttribute((const void*)(mlp_kernel<SINTERC, false>),
                              hipFuncAttributeMaxDynamicSharedMemorySize, MLP_LDS_S);

    // weight pre-pass: fp32 [R][C] -> bf16 [C][R]
    transpose_cvt_kernel<<<dim3(INTERC/64, DIMC/64, EC), 256, 0, stream>>>(w1, w1t, DIMC, INTERC);
    transpose_cvt_kernel<<<dim3(INTERC/64, DIMC/64, EC), 256, 0, stream>>>(w3, w3t, DIMC, INTERC);
    transpose_cvt_kernel<<<dim3(DIMC/64, INTERC/64, EC), 256, 0, stream>>>(w2, w2t, INTERC, DIMC);
    transpose_cvt_kernel<<<dim3(SINTERC/64, DIMC/64, 1), 256, 0, stream>>>(ws1, ws1t, DIMC, SINTERC);
    transpose_cvt_kernel<<<dim3(SINTERC/64, DIMC/64, 1), 256, 0, stream>>>(ws3, ws3t, DIMC, SINTERC);
    transpose_cvt_kernel<<<dim3(DIMC/64, SINTERC/64, 1), 256, 0, stream>>>(ws2, ws2t, SINTERC, DIMC);

    gate_kernel<<<TC / 64, 256, GATE_LDS, stream>>>(x, gate_w, scores);
    route_kernel<<<TC / 256, 256, 0, stream>>>(scores, idxb, wtsb);
    zero_cnt_kernel<<<1, 64, 0, stream>>>(cnt);
    pair_kernel<<<(TC * KC + 255) / 256, 256, 0, stream>>>(idxb, cnt, rowmap, pairpos);

    // routed experts -> eout (bf16)
    mlp_kernel<INTERC, true><<<dim3(CAPC / 64, EC), 512, MLP_LDS_D, stream>>>(
        x, w1t, w3t, w2t, rowmap, cnt, eout, nullptr);
    // shared expert -> d_out (fp32)
    mlp_kernel<SINTERC, false><<<dim3(TC / 64, 1), 512, MLP_LDS_S, stream>>>(
        x, ws1t, ws3t, ws2t, nullptr, nullptr, nullptr, out);

    combine_kernel<<<TC, 256, 0, stream>>>(idxb, pairpos, wtsb, eout, out);
}
